// LowRankValueCached_10943576670808
// MI455X (gfx1250) — 8-bit matrix operand order repaired and checked by reading, compiled, not run on hardware
//
#include <hip/hip_runtime.h>

typedef __attribute__((ext_vector_type(8))) int v8i;

#define BATCH   4
#define SEQ     4096
#define HIDDEN  4096
#define RANK    512
#define OUTF    4096
#define ROWS_V  (SEQ + 1)
#define ROWS_P  4128
#define QMAXF   127.0f
#define EPSQ    1e-8f

__global__ void lrvc_pack_a8(const int* __restrict__ A32, char* __restrict__ A8) {
    int i = blockIdx.x * blockDim.x + threadIdx.x;
    int4 v = ((const int4*)A32)[i];
    int packed = (v.x & 255) | ((v.y & 255) << 8) | ((v.z & 255) << 16) | (v.w << 24);
    ((int*)A8)[i] = packed;
}

__global__ void lrvc_gemv(const float* __restrict__ x, const float* __restrict__ Bw,
                          float* __restrict__ newlat) {
    int lane = threadIdx.x & 31;
    int wv   = threadIdx.x >> 5;
    int r    = blockIdx.x * 8 + wv;
    int b    = blockIdx.y;
    const float* xb = x  + (size_t)b * HIDDEN;
    const float* wr = Bw + (size_t)r * HIDDEN;
    float s = 0.f;
    for (int h = lane; h < HIDDEN; h += 32) s += xb[h] * wr[h];
    #pragma unroll
    for (int m = 16; m >= 1; m >>= 1) s += __shfl_xor(s, m, 32);
    if (lane == 0) newlat[b * RANK + r] = s;
}

__global__ void lrvc_quant(const float* __restrict__ cached, const float* __restrict__ newlat,
                           char* __restrict__ q, float* __restrict__ scales,
                           float* __restrict__ lat_out) {
    int lane = threadIdx.x & 31;
    int wv   = threadIdx.x >> 5;
    int s    = blockIdx.x * 8 + wv;
    int b    = blockIdx.y;

    float4 v[4];
    if (s < SEQ) {
        const float4* src = (const float4*)(cached + ((size_t)b * SEQ + s) * RANK);
        #pragma unroll
        for (int i = 0; i < 4; ++i) v[i] = src[i * 32 + lane];
    } else if (s == SEQ) {
        const float4* src = (const float4*)(newlat + (size_t)b * RANK);
        #pragma unroll
        for (int i = 0; i < 4; ++i) v[i] = src[i * 32 + lane];
    } else {
        #pragma unroll
        for (int i = 0; i < 4; ++i) v[i] = make_float4(0.f, 0.f, 0.f, 0.f);
    }

    float m = 0.f;
    #pragma unroll
    for (int i = 0; i < 4; ++i) {
        m = fmaxf(m, fabsf(v[i].x)); m = fmaxf(m, fabsf(v[i].y));
        m = fmaxf(m, fabsf(v[i].z)); m = fmaxf(m, fabsf(v[i].w));
    }
    #pragma unroll
    for (int off = 16; off >= 1; off >>= 1) m = fmaxf(m, __shfl_xor(m, off, 32));

    float amax  = fmaxf(m, EPSQ);
    float scale = amax / QMAXF;
    float inv   = QMAXF / amax;

    int* qrow = (int*)(q + ((size_t)b * ROWS_P + s) * RANK);
    #pragma unroll
    for (int i = 0; i < 4; ++i) {
        int q0 = min(127, max(-128, __float2int_rn(v[i].x * inv)));
        int q1 = min(127, max(-128, __float2int_rn(v[i].y * inv)));
        int q2 = min(127, max(-128, __float2int_rn(v[i].z * inv)));
        int q3 = min(127, max(-128, __float2int_rn(v[i].w * inv)));
        qrow[i * 32 + lane] = (q0 & 255) | ((q1 & 255) << 8) | ((q2 & 255) << 16) | (q3 << 24);
    }
    if (lane == 0) scales[b * ROWS_P + s] = scale;

    if (s < ROWS_V) {
        float4* dst = (float4*)(lat_out + ((size_t)b * ROWS_V + s) * RANK);
        #pragma unroll
        for (int i = 0; i < 4; ++i) dst[i * 32 + lane] = v[i];
    }
}

__global__ void lrvc_gemm_iu8(const char* __restrict__ q, const char* __restrict__ A8,
                              const float* __restrict__ scales,
                              const float* __restrict__ Ascale,
                              float* __restrict__ out) {
    int lane     = threadIdx.x & 31;
    int wv       = threadIdx.x >> 5;
    int colBase  = (blockIdx.x * 8 + wv) * 64;
    int rowBase  = blockIdx.y * 32;
    int b        = blockIdx.z;
    int half     = lane >> 4;
    int l15      = lane & 15;

    const char* qbase = q  + ((size_t)b * ROWS_P + rowBase + l15) * RANK;
    const char* bbase = A8 + ((size_t)(colBase + l15)) * RANK;

    v8i acc[2][4];
    #pragma unroll
    for (int t = 0; t < 2; ++t)
        #pragma unroll
        for (int u = 0; u < 4; ++u)
            acc[t][u] = (v8i){0, 0, 0, 0, 0, 0, 0, 0};

    for (int kk = 0; kk < 8; ++kk) {
        int k0 = kk * 64;
        v8i afrag[2], bfrag[4];
        #pragma unroll
        for (int t = 0; t < 2; ++t) {
            const char* p = qbase + t * (16 * RANK) + k0 + half * 16;
            int4 a0 = *(const int4*)(p);
            int4 a1 = *(const int4*)(p + 32);
            afrag[t] = (v8i){a0.x, a0.y, a0.z, a0.w, a1.x, a1.y, a1.z, a1.w};
        }
        #pragma unroll
        for (int u = 0; u < 4; ++u) {
            const char* p = bbase + u * (16 * RANK) + k0 + half * 16;
            int4 b0 = *(const int4*)(p);
            int4 b1 = *(const int4*)(p + 32);
            bfrag[u] = (v8i){b0.x, b0.y, b0.z, b0.w, b1.x, b1.y, b1.z, b1.w};
        }
        #pragma unroll
        for (int t = 0; t < 2; ++t)
            #pragma unroll
            for (int u = 0; u < 4; ++u)
                acc[t][u] = __builtin_amdgcn_wmma_i32_16x16x64_iu8(
                    true, afrag[t], true, bfrag[u], acc[t][u], false, false);
    }

    float cs[4];
    #pragma unroll
    for (int u = 0; u < 4; ++u) cs[u] = Ascale[colBase + u * 16 + l15];

    #pragma unroll
    for (int t = 0; t < 2; ++t) {
        #pragma unroll
        for (int r = 0; r < 8; ++r) {
            int row = rowBase + t * 16 + half * 8 + r;
            if (row < ROWS_V) {
                float rs = scales[b * ROWS_P + row];
                float* orow = out + ((size_t)b * ROWS_V + row) * OUTF + colBase + l15;
                #pragma unroll
                for (int u = 0; u < 4; ++u)
                    orow[u * 16] = (float)acc[t][u][r] * rs * cs[u];
            }
        }
    }
}

extern "C" void kernel_launch(void* const* d_in, const int* in_sizes, int n_in,
                              void* d_out, int out_size, void* d_ws, size_t ws_size,
                              hipStream_t stream) {
    const float* new_x  = (const float*)d_in[0];
    const float* cached = (const float*)d_in[1];
    const float* Bw     = (const float*)d_in[2];
    const int*   A32    = (const int*)  d_in[3];
    const float* Ascale = (const float*)d_in[4];

    float* out     = (float*)d_out;
    float* lat_out = out + (size_t)BATCH * ROWS_V * OUTF;

    char*  ws     = (char*)d_ws;
    char*  A8     = ws;
    float* newlat = (float*)(ws + 2097152);
    float* scales = (float*)(ws + 2097152 + 8192);
    char*  q      = ws + 2097152 + 8192 + 66048;

    lrvc_pack_a8<<<dim3((OUTF * RANK / 4) / 256), dim3(256), 0, stream>>>(A32, A8);

    lrvc_gemv<<<dim3(RANK / 8, BATCH), dim3(256), 0, stream>>>(new_x, Bw, newlat);

    lrvc_quant<<<dim3(ROWS_P / 8, BATCH), dim3(256), 0, stream>>>(cached, newlat, q,
                                                                  scales, lat_out);

    lrvc_gemm_iu8<<<dim3(OUTF / 64 / 8, ROWS_P / 32, BATCH), dim3(256), 0, stream>>>(
        q, A8, scales, Ascale, out);
}
